// NeuroHorizon_3281355014882
// MI455X (gfx1250) — compile-verified, matrix operand order checked by reading, not run on hardware
//
#include <hip/hip_runtime.h>
#include <hip/hip_bf16.h>
#include <math.h>

// ---------------- types ----------------
typedef __bf16 bf16;
typedef __attribute__((ext_vector_type(16))) __bf16 v16bf;
typedef __attribute__((ext_vector_type(8)))  __bf16 v8bf;
typedef __attribute__((ext_vector_type(4)))  __bf16 v4bf;
typedef __attribute__((ext_vector_type(8)))  float  v8f;
typedef __attribute__((ext_vector_type(4)))  float  v4f;

#define DIMC   512
#define DHC    64
#define NB     8          // batch
#define NIN    4096
#define NLAT   1120
#define TBINS  12
#define NUNITS 256

// WMMA fragment loader: A/B 16-bit layout (ISA 7.12.2): lane row = lane&15,
// K chunk base = (lane&16)?8:0; elements [kb..kb+7] and [kb+16..kb+23].
__device__ __forceinline__ v16bf ld_frag(const bf16* row, int kb) {
  v8bf lo = *(const v8bf*)(row + kb);
  v8bf hi = *(const v8bf*)(row + kb + 16);
  v16bf r;
#pragma unroll
  for (int i = 0; i < 8; i++) { r[i] = lo[i]; r[i + 8] = hi[i]; }
  return r;
}

__device__ __forceinline__ v8f wmma_bf16(v16bf a, v16bf b, v8f c) {
  return __builtin_amdgcn_wmma_f32_16x16x32_bf16(false, a, false, b, (short)0, c, false, false);
}

__device__ __forceinline__ float gelu_exact(float x) {
  return 0.5f * x * (1.0f + erff(x * 0.70710678118654752f));
}

// gfx1250 async global->LDS copy (ASYNCcnt-tracked, no VGPR data).
// Generic LDS pointer: low 32 bits are the LDS byte offset (aperture scheme).
__device__ __forceinline__ void async_copy_b128(const void* gptr, void* lptr) {
  unsigned l = (unsigned)(unsigned long long)(uintptr_t)lptr;
  asm volatile("global_load_async_to_lds_b128 %0, %1, off"
               : : "v"(l), "v"(gptr) : "memory");
}
__device__ __forceinline__ void wait_async0() {
  asm volatile("s_wait_asynccnt 0x0" : : : "memory");
}

// ---------------- weight convert+transpose: W[K][N] f32 -> Wt[N][K] bf16 ----
__global__ void k_wt(const float* __restrict__ W, bf16* __restrict__ Wt, int K, int N) {
  long long idx = (long long)blockIdx.x * blockDim.x + threadIdx.x;
  long long tot = (long long)K * N;
  if (idx >= tot) return;
  int n = (int)(idx % N);
  int k = (int)(idx / N);
  Wt[(size_t)n * K + k] = (bf16)W[(size_t)k * N + n];
}

// ---------------- embeddings / gathers ----------------
__global__ void k_embed_in(const int* __restrict__ uidx, const int* __restrict__ tt,
                           const float* __restrict__ uemb, const float* __restrict__ temb,
                           float* __restrict__ out, long long tot) {
  long long idx = (long long)blockIdx.x * blockDim.x + threadIdx.x;
  if (idx >= tot) return;
  long long tok = idx >> 9; int c = (int)(idx & 511);
  int u = uidx[tok], t = tt[tok];
  out[idx] = uemb[(size_t)u * DIMC + c] + temb[(size_t)t * DIMC + c];
}

__global__ void k_gather(const int* __restrict__ gidx, const float* __restrict__ emb,
                         float* __restrict__ out, long long tot) {
  long long idx = (long long)blockIdx.x * blockDim.x + threadIdx.x;
  if (idx >= tot) return;
  long long tok = idx >> 9; int c = (int)(idx & 511);
  out[idx] = emb[(size_t)gidx[tok] * DIMC + c];
}

__global__ void k_bcast_bins(const float* __restrict__ bin, float* __restrict__ out) {
  long long idx = (long long)blockIdx.x * blockDim.x + threadIdx.x;
  long long tot = (long long)NB * TBINS * DIMC;
  if (idx >= tot) return;
  long long tok = idx >> 9; int c = (int)(idx & 511);
  int t = (int)(tok % TBINS);
  out[idx] = bin[(size_t)t * DIMC + c];
}

__global__ void k_add(float* __restrict__ y, const float* __restrict__ x, long long n) {
  long long idx = (long long)blockIdx.x * blockDim.x + threadIdx.x;
  if (idx < n) y[idx] += x[idx];
}

// ---------------- layernorm: one wave32 per 512-wide row ----------------
__global__ __launch_bounds__(128) void k_ln(const float* __restrict__ X,
                                            const float* __restrict__ w, const float* __restrict__ b,
                                            float* __restrict__ Y, int rows) {
  int wid = threadIdx.x >> 5, lane = threadIdx.x & 31;
  int row = blockIdx.x * 4 + wid;
  if (row >= rows) return;
  const float* x = X + (size_t)row * DIMC;
  float v[16], s = 0.f, s2 = 0.f;
#pragma unroll
  for (int i = 0; i < 16; i++) { v[i] = x[lane + i * 32]; s += v[i]; s2 += v[i] * v[i]; }
#pragma unroll
  for (int off = 1; off < 32; off <<= 1) { s += __shfl_xor(s, off, 32); s2 += __shfl_xor(s2, off, 32); }
  float mean = s * (1.f / DIMC);
  float var = s2 * (1.f / DIMC) - mean * mean;
  float rstd = rsqrtf(var + 1e-5f);
  float* y = Y + (size_t)row * DIMC;
#pragma unroll
  for (int i = 0; i < 16; i++) { int c = lane + i * 32; y[c] = (v[i] - mean) * rstd * w[c] + b[c]; }
}

// ---------------- GEMM: C[M][N] = A[M][K](f32) @ Wt[N][K](bf16) + bias ------
// 64x64 block, BK=32, 4 waves each 32x32 (2x2 WMMA tiles).
// B tile staged with GLOBAL_LOAD_ASYNC_TO_LDS_B128; next A tile prefetched.
#define LDSA 40
__global__ __launch_bounds__(128, 1) void k_gemm(const float* __restrict__ A,
                                                 const bf16* __restrict__ Wt,
                                                 const float* __restrict__ bias,
                                                 float* __restrict__ C, int M, int N, int K) {
  __shared__ bf16 sA[64 * LDSA];
  __shared__ bf16 sB[64 * LDSA];
  int tid = threadIdx.x, lane = tid & 31, wid = tid >> 5;
  int wm = wid >> 1, wn = wid & 1;
  int m0 = blockIdx.y * 64, n0 = blockIdx.x * 64;
  int nlo = lane & 15, kb = (lane & 16) ? 8 : 0, rh = (lane & 16) ? 8 : 0;
  v8f acc[2][2] = {};
  for (int kk = 0; kk < K; kk += 32) {
    // stage B tile asynchronously: 64x32 bf16 = 4KB, 2 x b128 per thread
#pragma unroll
    for (int it = 0; it < 2; it++) {
      int e = tid * 8 + it * 1024;
      int r = e >> 5, c = e & 31;
      async_copy_b128(Wt + (size_t)(n0 + r) * K + kk + c, sB + r * LDSA + c);
    }
    // stage A (fp32 -> bf16); clamped-address load + select (no exec branch)
#pragma unroll
    for (int it = 0; it < 4; it++) {
      int e = tid * 4 + it * 512;
      int r = e >> 5, c = e & 31;
      int gm = m0 + r;
      bool ok = gm < M;
      int gmc = ok ? gm : m0;
      v4f av = *(const v4f*)(A + (size_t)gmc * K + kk + c);
      v4bf bv;
#pragma unroll
      for (int u = 0; u < 4; u++) bv[u] = (bf16)(ok ? av[u] : 0.f);
      *(v4bf*)(sA + r * LDSA + c) = bv;
    }
    // prefetch next A tile (global_prefetch_b8)
    if (kk + 32 < K) {
      int e = tid * 4;
      int r = e >> 5, c = e & 31;
      int gm = m0 + r;
      __builtin_prefetch(A + (size_t)(gm < M ? gm : m0) * K + kk + 32 + c, 0, 0);
    }
    wait_async0();
    __syncthreads();
    v16bf a0 = ld_frag(sA + (wm * 32 + nlo) * LDSA, kb);
    v16bf a1 = ld_frag(sA + (wm * 32 + 16 + nlo) * LDSA, kb);
    v16bf b0 = ld_frag(sB + (wn * 32 + nlo) * LDSA, kb);
    v16bf b1 = ld_frag(sB + (wn * 32 + 16 + nlo) * LDSA, kb);
    acc[0][0] = wmma_bf16(a0, b0, acc[0][0]);
    acc[0][1] = wmma_bf16(a0, b1, acc[0][1]);
    acc[1][0] = wmma_bf16(a1, b0, acc[1][0]);
    acc[1][1] = wmma_bf16(a1, b1, acc[1][1]);
    __syncthreads();
  }
#pragma unroll
  for (int mi = 0; mi < 2; mi++) {
#pragma unroll
    for (int ni = 0; ni < 2; ni++) {
      int col = n0 + wn * 32 + ni * 16 + nlo;
      float bv = bias ? bias[col] : 0.f;
#pragma unroll
      for (int j = 0; j < 8; j++) {
        int m = m0 + wm * 32 + mi * 16 + rh + j;
        if (m < M) C[(size_t)m * N + col] = acc[mi][ni][j] + bv;
      }
    }
  }
}

// ---------------- RoPE ----------------
// pairs j=0..31; only j<16 rotate: invf = 2*pi / (T_MIN * R^(j/16)), R=T_MAX/T_MIN
__device__ __forceinline__ void rope_cs(float t, int j, float& c, float& s) {
  c = 1.f; s = 0.f;
  if (j < 16) {
    float invf = 6.283185307179586f / (0.0001f * powf(20627.0f, (float)j * (1.f / 16.f)));
    float ang = t * invf;
    c = cosf(ang); s = sinf(ang);
  }
}

// scatter: X[b][n][seg + h*64 + :64] -> Out[b*H+h][n][64] with rotary applied
__global__ void k_rope_scatter(const float* __restrict__ X, int ldx, int seg,
                               const float* __restrict__ ts, float* __restrict__ Out,
                               int Nt, int H) {
  long long idx = (long long)blockIdx.x * blockDim.x + threadIdx.x;
  long long tot = (long long)NB * Nt * H * 32;
  if (idx >= tot) return;
  int j = (int)(idx & 31); long long r = idx >> 5;
  int h = (int)(r % H); r /= H;
  int n = (int)(r % Nt); int b = (int)(r / Nt);
  float t = ts[(size_t)b * Nt + n];
  float c, s; rope_cs(t, j, c, s);
  const float* src = X + ((size_t)b * Nt + n) * ldx + seg + h * DHC + 2 * j;
  float x0 = src[0], x1 = src[1];
  float* dst = Out + (((size_t)(b * H + h) * Nt + n) * DHC) + 2 * j;
  dst[0] = x0 * c - x1 * s;
  dst[1] = x1 * c + x0 * s;
}

// merge: O[b*H+h][n][64] -> Out[b][n][H*64] with inverse rotary
__global__ void k_rope_merge(const float* __restrict__ O, const float* __restrict__ ts,
                             float* __restrict__ Out, int Nt, int H) {
  long long idx = (long long)blockIdx.x * blockDim.x + threadIdx.x;
  long long tot = (long long)NB * Nt * H * 32;
  if (idx >= tot) return;
  int j = (int)(idx & 31); long long r = idx >> 5;
  int h = (int)(r % H); r /= H;
  int n = (int)(r % Nt); int b = (int)(r / Nt);
  float t = ts[(size_t)b * Nt + n];
  float c, s; rope_cs(t, j, c, s);
  const float* src = O + (((size_t)(b * H + h) * Nt + n) * DHC) + 2 * j;
  float x0 = src[0], x1 = src[1];
  float* dst = Out + ((size_t)b * Nt + n) * (H * DHC) + h * DHC + 2 * j;
  dst[0] = x0 * c + x1 * s;   // apply_rotary(-f)
  dst[1] = x1 * c - x0 * s;
}

// ---------------- flash attention (DH=64), wave owns 16 q rows -------------
#define LK 72
__global__ __launch_bounds__(128, 1) void k_attn(const float* __restrict__ Q,
                                                 const float* __restrict__ Kp,
                                                 const float* __restrict__ Vp,
                                                 float* __restrict__ O,
                                                 const unsigned char* __restrict__ mask,
                                                 int H, int Nq, int Nk, int causal, float scale) {
  __shared__ bf16 sK[64 * LK];
  __shared__ bf16 sVT[64 * LK];
  __shared__ bf16 sP[4][16 * LK];
  int tid = threadIdx.x, lane = tid & 31, wid = tid >> 5;
  int bh = blockIdx.y;
  int b = bh / H;
  int nlo = lane & 15, kb = (lane & 16) ? 8 : 0, rh = (lane & 16) ? 8 : 0;
  int qbase = blockIdx.x * 64 + wid * 16;
  const float* Qb = Q + (size_t)bh * Nq * DHC;
  const float* Kb = Kp + (size_t)bh * Nk * DHC;
  const float* Vb = Vp + (size_t)bh * Nk * DHC;

  // stage this wave's 16 q rows into its private sP region (bf16)
  {
    int r = lane & 15, dh = (lane >> 4) * 32;
    int q = qbase + r;
    bool ok = q < Nq;
    int qc = ok ? q : (Nq - 1);
#pragma unroll
    for (int i = 0; i < 8; i++) {
      int d = dh + i * 4;
      v4f qv = *(const v4f*)(Qb + (size_t)qc * DHC + d);
      v4bf bv;
#pragma unroll
      for (int u = 0; u < 4; u++) bv[u] = (bf16)(ok ? qv[u] : 0.f);
      *(v4bf*)(&sP[wid][r * LK + d]) = bv;
    }
  }
  __syncthreads();
  v16bf qf0 = ld_frag(&sP[wid][nlo * LK + 0], kb);
  v16bf qf1 = ld_frag(&sP[wid][nlo * LK + 32], kb);
  __syncthreads();

  float mrow[8], lrow[8];
#pragma unroll
  for (int j = 0; j < 8; j++) { mrow[j] = -3e38f; lrow[j] = 0.f; }
  v8f oacc[4] = {};

  int ntile = (Nk + 63) >> 6;
  for (int kt = 0; kt < ntile; kt++) {
    int k0 = kt * 64;
    // stage K (row-major) and V^T; clamped loads, zero-select for padding
#pragma unroll
    for (int it = 0; it < 8; it++) {
      int e = tid * 4 + it * 512;
      int r = e >> 6, d = e & 63;
      int kp = k0 + r;
      bool ok = kp < Nk;
      int kpc = ok ? kp : k0;
      v4f kv = *(const v4f*)(Kb + (size_t)kpc * DHC + d);
      v4f vv = *(const v4f*)(Vb + (size_t)kpc * DHC + d);
      v4bf bv;
#pragma unroll
      for (int u = 0; u < 4; u++) bv[u] = (bf16)(ok ? kv[u] : 0.f);
      *(v4bf*)(&sK[r * LK + d]) = bv;
#pragma unroll
      for (int u = 0; u < 4; u++) sVT[(d + u) * LK + r] = (bf16)(ok ? vv[u] : 0.f);
    }
    __syncthreads();

    // S = Q K^T
    v8f sacc[4] = {};
#pragma unroll
    for (int ni = 0; ni < 4; ni++) {
      v16bf bk0 = ld_frag(&sK[(ni * 16 + nlo) * LK + 0], kb);
      sacc[ni] = wmma_bf16(qf0, bk0, sacc[ni]);
      v16bf bk1 = ld_frag(&sK[(ni * 16 + nlo) * LK + 32], kb);
      sacc[ni] = wmma_bf16(qf1, bk1, sacc[ni]);
    }
    // scale + mask (padding / input mask / causal)
#pragma unroll
    for (int ni = 0; ni < 4; ni++) {
      int kvp = k0 + ni * 16 + nlo;
      bool kvok = (kvp < Nk) && (!mask || (mask[(size_t)b * Nk + (kvp < Nk ? kvp : 0)] != 0));
#pragma unroll
      for (int j = 0; j < 8; j++) {
        int qp = qbase + rh + j;
        bool ok = kvok && (!causal || kvp <= qp);
        float sv = sacc[ni][j] * scale;
        sacc[ni][j] = ok ? sv : -1e9f;
      }
    }
    // online softmax (16-lane halves hold the 16 columns of each row)
#pragma unroll
    for (int j = 0; j < 8; j++) {
      float mx = fmaxf(fmaxf(sacc[0][j], sacc[1][j]), fmaxf(sacc[2][j], sacc[3][j]));
#pragma unroll
      for (int off = 1; off < 16; off <<= 1) mx = fmaxf(mx, __shfl_xor(mx, off, 32));
      float mnew = fmaxf(mrow[j], mx);
      float corr = __expf(mrow[j] - mnew);
      mrow[j] = mnew;
      float rs = 0.f;
#pragma unroll
      for (int ni = 0; ni < 4; ni++) {
        float pv = __expf(sacc[ni][j] - mnew);
        sacc[ni][j] = pv; rs += pv;
      }
#pragma unroll
      for (int off = 1; off < 16; off <<= 1) rs += __shfl_xor(rs, off, 32);
      lrow[j] = lrow[j] * corr + rs;
#pragma unroll
      for (int ni = 0; ni < 4; ni++) oacc[ni][j] *= corr;
    }
    // P -> LDS (bf16, A layout staging)
#pragma unroll
    for (int ni = 0; ni < 4; ni++)
#pragma unroll
      for (int j = 0; j < 8; j++)
        sP[wid][(rh + j) * LK + ni * 16 + nlo] = (bf16)sacc[ni][j];
    __syncthreads();
    // O += P V
    v16bf pf0 = ld_frag(&sP[wid][nlo * LK + 0], kb);
    v16bf pf1 = ld_frag(&sP[wid][nlo * LK + 32], kb);
#pragma unroll
    for (int ni = 0; ni < 4; ni++) {
      v16bf bv0 = ld_frag(&sVT[(ni * 16 + nlo) * LK + 0], kb);
      oacc[ni] = wmma_bf16(pf0, bv0, oacc[ni]);
      v16bf bv1 = ld_frag(&sVT[(ni * 16 + nlo) * LK + 32], kb);
      oacc[ni] = wmma_bf16(pf1, bv1, oacc[ni]);
    }
    __syncthreads();
  }
#pragma unroll
  for (int j = 0; j < 8; j++) {
    int qp = qbase + rh + j;
    if (qp < Nq) {
      float inv = 1.f / lrow[j];
#pragma unroll
      for (int ni = 0; ni < 4; ni++)
        O[((size_t)bh * Nq + qp) * DHC + ni * 16 + nlo] = oacc[ni][j] * inv;
    }
  }
}

// ---------------- GLU: h = a * gelu(g), ag = [M][4096] -------------------
__global__ void k_glu(const float* __restrict__ AG, float* __restrict__ Hb, long long M) {
  long long idx = (long long)blockIdx.x * blockDim.x + threadIdx.x;
  long long tot = M * 2048;
  if (idx >= tot) return;
  long long m = idx >> 11; int c = (int)(idx & 2047);
  float a = AG[m * 4096 + c];
  float g = AG[m * 4096 + 2048 + c];
  Hb[idx] = a * gelu_exact(g);
}

// ---------------- head: out[b,t,n] = w2 . gelu(hb[b,t]+hu[b,n]+b1) + b2 ----
__global__ __launch_bounds__(256) void k_head(const float* __restrict__ HB, const float* __restrict__ HU,
                                              const float* __restrict__ b1, const float* __restrict__ w2,
                                              const float* __restrict__ b2, float* __restrict__ out) {
  int wid = threadIdx.x >> 5, lane = threadIdx.x & 31;
  int gid = blockIdx.x * 8 + wid;
  if (gid >= NB * TBINS * NUNITS) return;
  int n = gid & 255;
  int r = gid >> 8;
  int t = r % TBINS, b = r / TBINS;
  const float* hb = HB + (size_t)(b * TBINS + t) * DIMC;
  const float* hu = HU + (size_t)(b * NUNITS + n) * DIMC;
  float acc = 0.f;
#pragma unroll
  for (int i = 0; i < 16; i++) {
    int c = lane + i * 32;
    float h = hb[c] + hu[c] + b1[c];
    acc += gelu_exact(h) * w2[c];
  }
#pragma unroll
  for (int off = 1; off < 32; off <<= 1) acc += __shfl_xor(acc, off, 32);
  if (lane == 0) out[(size_t)gid] = acc + b2[0];
}

// ================= host orchestration =================
// param leaf indices (alphabetical pytree order after the 8 data inputs)
#define PI_BIN_EMB  8
#define PI_DEC0     9   // ca(8): bo,ln_c_b,ln_c_w,ln_q_b,ln_q_w,wkv,wo,wq ; ff(6): b1,b2,ln_b,ln_w,w1,w2 ; sa(5): bo,ln_b,ln_w,wo,wqkv
#define PI_DEC1     28
#define PI_ENC_ATN  47
#define PI_ENC_FFN  55
#define PI_HEAD     61  // b1,b2,w2,wb,wu
#define PI_LAT_EMB  66
#define PI_PROC0    67  // ff(6), sa(5)
#define PI_PROC1    78
#define PI_TT_EMB   89
#define PI_UNIT_EMB 90

static inline long long cdiv(long long a, long long b) { return (a + b - 1) / b; }

extern "C" void kernel_launch(void* const* d_in, const int* in_sizes, int n_in,
                              void* d_out, int out_size, void* d_ws, size_t ws_size,
                              hipStream_t stream) {
  (void)in_sizes; (void)n_in; (void)out_size; (void)ws_size;

  const int*           in_uidx = (const int*)d_in[0];
  const float*         in_ts   = (const float*)d_in[1];
  const int*           in_tt   = (const int*)d_in[2];
  const unsigned char* in_mask = (const unsigned char*)d_in[3];
  const int*           lat_idx = (const int*)d_in[4];
  const float*         lat_ts  = (const float*)d_in[5];
  const float*         bin_ts  = (const float*)d_in[6];
  const int*           tgt_idx = (const int*)d_in[7];
  auto p = [&](int i) { return (const float*)d_in[i]; };

  size_t off = 0;
  auto alloc = [&](size_t bytes) -> void* {
    void* r = (char*)d_ws + off;
    off += (bytes + 255) & ~(size_t)255;
    return r;
  };
  auto allocf = [&](size_t n) { return (float*)alloc(n * sizeof(float)); };

  auto prepw = [&](int idx, int Kd, int Nd) -> bf16* {
    bf16* dst = (bf16*)alloc((size_t)Kd * Nd * sizeof(bf16));
    long long tot = (long long)Kd * Nd;
    k_wt<<<(unsigned)cdiv(tot, 256), 256, 0, stream>>>(p(idx), dst, Kd, Nd);
    return dst;
  };

  // ---- weights -> bf16 transposed ----
  bf16* e_wq  = prepw(PI_ENC_ATN + 7, 512, 128);
  bf16* e_wkv = prepw(PI_ENC_ATN + 5, 512, 256);
  bf16* e_wo  = prepw(PI_ENC_ATN + 6, 128, 512);
  bf16* e_w1  = prepw(PI_ENC_FFN + 4, 512, 4096);
  bf16* e_w2  = prepw(PI_ENC_FFN + 5, 2048, 512);
  bf16 *pr_wqkv[2], *pr_wo[2], *pr_w1[2], *pr_w2[2];
  for (int i = 0; i < 2; i++) {
    int base = i ? PI_PROC1 : PI_PROC0;
    pr_w1[i]   = prepw(base + 4, 512, 4096);
    pr_w2[i]   = prepw(base + 5, 2048, 512);
    pr_wo[i]   = prepw(base + 6 + 3, 512, 512);
    pr_wqkv[i] = prepw(base + 6 + 4, 512, 1536);
  }
  bf16 *d_swqkv[2], *d_swo[2], *d_cwq[2], *d_cwkv[2], *d_cwo[2], *d_w1[2], *d_w2[2];
  for (int i = 0; i < 2; i++) {
    int base = i ? PI_DEC1 : PI_DEC0;
    d_cwkv[i]  = prepw(base + 5, 512, 256);
    d_cwo[i]   = prepw(base + 6, 128, 512);
    d_cwq[i]   = prepw(base + 7, 512, 128);
    d_w1[i]    = prepw(base + 8 + 4, 512, 4096);
    d_w2[i]    = prepw(base + 8 + 5, 2048, 512);
    d_swo[i]   = prepw(base + 14 + 3, 512, 512);
    d_swqkv[i] = prepw(base + 14 + 4, 512, 1536);
  }
  bf16* h_wb = prepw(PI_HEAD + 3, 512, 512);
  bf16* h_wu = prepw(PI_HEAD + 4, 512, 512);

  // ---- activation buffers ----
  const int ML = NB * NLAT;   // 8960
  float* Xin = allocf((size_t)NB * NIN * DIMC);          // 16.8M
  float* Lat = allocf((size_t)ML * DIMC);
  float* T0  = allocf((size_t)ML * DIMC);
  float* T1  = allocf((size_t)ML * DIMC);
  float* QKV = allocf((size_t)ML * 1536);
  float* KV  = allocf((size_t)NB * NIN * 256);
  float* Qh  = allocf((size_t)NB * 8 * NLAT * DHC);
  float* Kh  = allocf((size_t)NB * 8 * NLAT * DHC);
  float* Vh  = allocf((size_t)NB * 8 * NLAT * DHC);
  float* Oh  = allocf((size_t)NB * 8 * NLAT * DHC);
  float* Mrg = allocf((size_t)ML * DIMC);
  float* AG  = allocf((size_t)ML * 4096);
  float* Hb  = allocf((size_t)ML * 2048);
  float* Xd  = allocf((size_t)NB * TBINS * DIMC);
  float* HBv = allocf((size_t)NB * TBINS * DIMC);
  float* UE  = allocf((size_t)NB * NUNITS * DIMC);
  float* HUv = allocf((size_t)NB * NUNITS * DIMC);

  // ---- launch helpers ----
  auto ln = [&](const float* x, int wi, int bi, float* y, int rows) {
    k_ln<<<(unsigned)cdiv(rows, 4), 128, 0, stream>>>(x, p(wi), p(bi), y, rows);
  };
  auto gemm = [&](const float* A, const bf16* Wt, const float* bias, float* C, int M, int N, int K) {
    dim3 g(N / 64, (unsigned)cdiv(M, 64));
    k_gemm<<<g, 128, 0, stream>>>(A, Wt, bias, C, M, N, K);
  };
  auto addv = [&](float* y, const float* x, long long n) {
    k_add<<<(unsigned)cdiv(n, 256), 256, 0, stream>>>(y, x, n);
  };
  auto rope_s = [&](const float* X, int ldx, int seg, const float* ts, float* Out, int Nt, int H) {
    long long tot = (long long)NB * Nt * H * 32;
    k_rope_scatter<<<(unsigned)cdiv(tot, 256), 256, 0, stream>>>(X, ldx, seg, ts, Out, Nt, H);
  };
  auto rope_m = [&](const float* O, const float* ts, float* Out, int Nt, int H) {
    long long tot = (long long)NB * Nt * H * 32;
    k_rope_merge<<<(unsigned)cdiv(tot, 256), 256, 0, stream>>>(O, ts, Out, Nt, H);
  };
  auto attn = [&](const float* Qp, const float* Kp, const float* Vp, float* Op,
                  const unsigned char* msk, int H, int Nq, int Nk, int causal) {
    dim3 g((unsigned)cdiv(Nq, 64), NB * H);
    k_attn<<<g, 128, 0, stream>>>(Qp, Kp, Vp, Op, msk, H, Nq, Nk, causal, 0.125f);
  };
  auto ffn = [&](float* x, int rows, int base, const bf16* w1t, const bf16* w2t) {
    ln(x, base + 3, base + 2, T0, rows);
    gemm(T0, w1t, p(base + 0), AG, rows, 4096, 512);
    long long tot = (long long)rows * 2048;
    k_glu<<<(unsigned)cdiv(tot, 256), 256, 0, stream>>>(AG, Hb, rows);
    gemm(Hb, w2t, p(base + 1), T1, rows, 512, 2048);
    addv(x, T1, (long long)rows * 512);
  };

  // ---- embeddings ----
  {
    long long tot = (long long)NB * NIN * DIMC;
    k_embed_in<<<(unsigned)cdiv(tot, 256), 256, 0, stream>>>(in_uidx, in_tt, p(PI_UNIT_EMB), p(PI_TT_EMB), Xin, tot);
    long long tl = (long long)ML * DIMC;
    k_gather<<<(unsigned)cdiv(tl, 256), 256, 0, stream>>>(lat_idx, p(PI_LAT_EMB), Lat, tl);
  }

  // ---- encoder cross-attention ----
  ln(Lat, PI_ENC_ATN + 4, PI_ENC_ATN + 3, T0, ML);            // ln_q
  gemm(T0, e_wq, nullptr, QKV, ML, 128, 512);                 // q
  ln(Xin, PI_ENC_ATN + 2, PI_ENC_ATN + 1, Xin, NB * NIN);     // ln_c (in place)
  gemm(Xin, e_wkv, nullptr, KV, NB * NIN, 256, 512);          // kv
  rope_s(QKV, 128, 0, lat_ts, Qh, NLAT, 2);
  rope_s(KV, 256, 0, in_ts, Kh, NIN, 2);
  rope_s(KV, 256, 128, in_ts, Vh, NIN, 2);
  attn(Qh, Kh, Vh, Oh, in_mask, 2, NLAT, NIN, 0);
  rope_m(Oh, lat_ts, Mrg, NLAT, 2);
  gemm(Mrg, e_wo, p(PI_ENC_ATN + 0), T1, ML, 512, 128);
  addv(Lat, T1, (long long)ML * 512);
  ffn(Lat, ML, PI_ENC_FFN, e_w1, e_w2);

  // ---- processor layers ----
  for (int i = 0; i < 2; i++) {
    int base = i ? PI_PROC1 : PI_PROC0;
    int sa = base + 6;
    ln(Lat, sa + 2, sa + 1, T0, ML);
    gemm(T0, pr_wqkv[i], nullptr, QKV, ML, 1536, 512);
    rope_s(QKV, 1536, 0, lat_ts, Qh, NLAT, 8);
    rope_s(QKV, 1536, 512, lat_ts, Kh, NLAT, 8);
    rope_s(QKV, 1536, 1024, lat_ts, Vh, NLAT, 8);
    attn(Qh, Kh, Vh, Oh, nullptr, 8, NLAT, NLAT, 0);
    rope_m(Oh, lat_ts, Mrg, NLAT, 8);
    gemm(Mrg, pr_wo[i], p(sa + 0), T1, ML, 512, 512);
    addv(Lat, T1, (long long)ML * 512);
    ffn(Lat, ML, base, pr_w1[i], pr_w2[i]);
  }

  // ---- decoder ----
  {
    long long tot = (long long)NB * TBINS * DIMC;
    k_bcast_bins<<<(unsigned)cdiv(tot, 256), 256, 0, stream>>>(p(PI_BIN_EMB), Xd);
  }
  const int MD = NB * TBINS; // 96
  for (int i = 0; i < 2; i++) {
    int base = i ? PI_DEC1 : PI_DEC0;
    int ca = base, ff = base + 8, sa = base + 14;
    // causal self-attn over bins
    ln(Xd, sa + 2, sa + 1, T0, MD);
    gemm(T0, d_swqkv[i], nullptr, QKV, MD, 1536, 512);
    rope_s(QKV, 1536, 0, bin_ts, Qh, TBINS, 8);
    rope_s(QKV, 1536, 512, bin_ts, Kh, TBINS, 8);
    rope_s(QKV, 1536, 1024, bin_ts, Vh, TBINS, 8);
    attn(Qh, Kh, Vh, Oh, nullptr, 8, TBINS, TBINS, 1);
    rope_m(Oh, bin_ts, Mrg, TBINS, 8);
    gemm(Mrg, d_swo[i], p(sa + 0), T1, MD, 512, 512);
    addv(Xd, T1, (long long)MD * 512);
    // cross-attn to latents
    ln(Xd, ca + 4, ca + 3, T0, MD);
    ln(Lat, ca + 2, ca + 1, Hb, ML);   // latent LN reuses Hb arena
    gemm(T0, d_cwq[i], nullptr, QKV, MD, 128, 512);
    gemm(Hb, d_cwkv[i], nullptr, KV, ML, 256, 512);
    rope_s(QKV, 128, 0, bin_ts, Qh, TBINS, 2);
    rope_s(KV, 256, 0, lat_ts, Kh, NLAT, 2);
    rope_s(KV, 256, 128, lat_ts, Vh, NLAT, 2);
    attn(Qh, Kh, Vh, Oh, nullptr, 2, TBINS, NLAT, 0);
    rope_m(Oh, bin_ts, Mrg, TBINS, 2);
    gemm(Mrg, d_cwo[i], p(ca + 0), T1, MD, 512, 128);
    addv(Xd, T1, (long long)MD * 512);
    // ffn
    ffn(Xd, MD, ff, d_w1[i], d_w2[i]);
  }

  // ---- head ----
  gemm(Xd, h_wb, nullptr, HBv, MD, 512, 512);
  {
    long long tl = (long long)NB * NUNITS * DIMC;
    k_gather<<<(unsigned)cdiv(tl, 256), 256, 0, stream>>>(tgt_idx, p(PI_UNIT_EMB), UE, tl);
  }
  gemm(UE, h_wu, nullptr, HUv, NB * NUNITS, 512, 512);
  {
    int waves = NB * TBINS * NUNITS;  // 24576 (one wave each)
    k_head<<<(unsigned)cdiv(waves, 8), 256, 0, stream>>>(HBv, HUv, p(PI_HEAD + 0), p(PI_HEAD + 2), p(PI_HEAD + 1), (float*)d_out);
  }
}
